// KAGNMoE_70866960384513
// MI455X (gfx1250) — compile-verified, matrix operand order checked by reading, not run on hardware
//
#include <hip/hip_runtime.h>
#include <hip/hip_bf16.h>
#include <math.h>

typedef __attribute__((ext_vector_type(16))) _Float16 v16h;
typedef __attribute__((ext_vector_type(8)))  _Float16 v8h;
typedef __attribute__((ext_vector_type(8)))  float    v8f;

#define BATCH 16
#define CIN   32
#define HW    192
#define NEXP  8
#define GCH   128              // 4 Gram bases * 32 input channels
#define TAPS  9
#define KTOT  (TAPS * GCH)     // 1152
#define WPAD  1160             // padded K stride (halfs) for weights in LDS
#define GPAD  136              // padded channel stride (halfs) for g in LDS
#define TILE  16
#define SROW  18               // TILE + 2 halo
#define NTHREADS 256           // 8 waves; wave w owns output rows 2w, 2w+1

union V16u { v16h v; v8h h[2]; };

__device__ __forceinline__ float siluf(float v) {
  return v / (1.0f + __expf(-v));
}

// ---------------------------------------------------------------------------
// Kernel 1: per-(b,c) mean over 192x192  ->  gmean[b*32+c]
// ---------------------------------------------------------------------------
__global__ __launch_bounds__(256) void kagn_colmean_kernel(
    const float* __restrict__ x, float* __restrict__ gmean) {
  const int c = blockIdx.x;
  const int b = blockIdx.y;
  const float4* p = (const float4*)(x + ((size_t)b * CIN + c) * (HW * HW));
  float s = 0.0f;
  for (int i = threadIdx.x; i < (HW * HW) / 4; i += 256) {
    float4 v = p[i];
    s += v.x + v.y + v.z + v.w;
  }
  __shared__ float red[256];
  red[threadIdx.x] = s;
  __syncthreads();
  for (int st = 128; st > 0; st >>= 1) {
    if (threadIdx.x < st) red[threadIdx.x] += red[threadIdx.x + st];
    __syncthreads();
  }
  if (threadIdx.x == 0)
    gmean[b * CIN + c] = red[0] * (1.0f / (float)(HW * HW));
}

// ---------------------------------------------------------------------------
// Kernel 2: softmax gate, top-2, per-batch scale, aux loss (tiny, serial)
// ---------------------------------------------------------------------------
__device__ __forceinline__ float cv2_8(const float* v) {
  float m = 0.0f;
  for (int e = 0; e < NEXP; ++e) m += v[e];
  m *= (1.0f / NEXP);
  float var = 0.0f;
  for (int e = 0; e < NEXP; ++e) { float d = v[e] - m; var += d * d; }
  var *= (1.0f / (NEXP - 1));           // ddof = 1
  return var / (m * m + 1e-10f);
}

__global__ void kagn_gate_kernel(const float* __restrict__ gmean,
                                 const float* __restrict__ wgate,
                                 float* __restrict__ scale,
                                 float* __restrict__ loss_out) {
  if (threadIdx.x != 0 || blockIdx.x != 0) return;
  float imp[NEXP];
  float loadc[NEXP];
  for (int e = 0; e < NEXP; ++e) { imp[e] = 0.0f; loadc[e] = 0.0f; }
  for (int b = 0; b < BATCH; ++b) {
    float logits[NEXP];
    float mx = -1e30f;
    for (int e = 0; e < NEXP; ++e) {
      float a = 0.0f;
      for (int c = 0; c < CIN; ++c) a += gmean[b * CIN + c] * wgate[c * NEXP + e];
      logits[e] = a;
      mx = fmaxf(mx, a);
    }
    float den = 0.0f;
    for (int e = 0; e < NEXP; ++e) { logits[e] = __expf(logits[e] - mx); den += logits[e]; }
    float inv = 1.0f / den;
    for (int e = 0; e < NEXP; ++e) logits[e] *= inv;
    // top-2 (first occurrence wins on ties, matching top_k)
    int i0 = 0;
    for (int e = 1; e < NEXP; ++e) if (logits[e] > logits[i0]) i0 = e;
    int i1 = -1;
    for (int e = 0; e < NEXP; ++e)
      if (e != i0 && (i1 < 0 || logits[e] > logits[i1])) i1 = e;
    float v0 = logits[i0], v1 = logits[i1];
    float dn = v0 + v1 + 1e-6f;
    float g0 = v0 / dn, g1 = v1 / dn;
    scale[b] = g0 + g1;
    imp[i0] += g0; imp[i1] += g1;
    loadc[i0] += (g0 > 0.0f) ? 1.0f : 0.0f;
    loadc[i1] += (g1 > 0.0f) ? 1.0f : 0.0f;
  }
  *loss_out = (cv2_8(imp) + cv2_8(loadc)) * 0.01f;
}

// ---------------------------------------------------------------------------
// Kernel 3: fused tanh -> Gram basis -> silu -> 3x3 conv (implicit GEMM,
// f16 WMMA with f32 accumulation) -> scale by per-batch gate sum.
// 256 threads = 8 waves; wave w computes output rows (ty0+2w, ty0+2w+1) so
// each B fragment is reused by two A fragments: 2 ds_load_b128 per WMMA.
// ---------------------------------------------------------------------------
__global__ __launch_bounds__(NTHREADS) void kagn_conv_kernel(
    const float* __restrict__ x, const float* __restrict__ poly,
    const float* __restrict__ betaw, const float* __restrict__ scale,
    float* __restrict__ out) {
  __shared__ _Float16 sW[32 * WPAD];              // Wt[o][k], k = tap*128 + jc
  __shared__ _Float16 sG[SROW * SROW * GPAD];     // g[row][col][128 ch]

  const int tx0 = blockIdx.x * TILE;
  const int ty0 = blockIdx.y * TILE;
  const int b   = blockIdx.z;
  const int tid = threadIdx.x;

  // ---- stage weights (fp32 -> f16), transposed to [o][k] ----
  for (int idx = tid; idx < 32 * KTOT; idx += NTHREADS) {
    int o   = idx / KTOT;
    int k   = idx - o * KTOT;
    int tap = k >> 7;                 // k / 128
    int jc  = k & 127;                // basis*32 + cin
    float w = poly[(size_t)(o * GCH + jc) * TAPS + tap];
    sW[o * WPAD + k] = (_Float16)w;
  }

  // ---- stage g tile (with halo): tanh -> Gram P0..P3 -> silu -> f16 ----
  const float beta2 = 2.25f * betaw[1];            // (3*1*1*3/4) * beta_w[1]
  const float beta3 = (300.0f / 9.0f) * betaw[2];  // (5*1*4*15/9) * beta_w[2]
  for (int idx = tid; idx < CIN * SROW * SROW; idx += NTHREADS) {
    int c   = idx / (SROW * SROW);
    int pos = idx - c * (SROW * SROW);
    int r   = pos / SROW;
    int col = pos - r * SROW;
    int gy = ty0 + r - 1;
    int gx = tx0 + col - 1;
    float p0 = 0.0f, p1 = 0.0f, p2 = 0.0f, p3 = 0.0f;
    if (gy >= 0 && gy < HW && gx >= 0 && gx < HW) {
      float t = tanhf(x[(((size_t)b * CIN + c) * HW + gy) * HW + gx]);
      float q2 = t * t - beta2;          // P2 = t*P1 - beta2*P0
      p0 = siluf(1.0f);
      p1 = siluf(t);
      p2 = siluf(q2);
      p3 = siluf(t * q2 - beta3 * t);    // P3 = t*P2 - beta3*P1
    }
    _Float16* gp = sG + (r * SROW + col) * GPAD;
    gp[c]      = (_Float16)p0;
    gp[32 + c] = (_Float16)p1;
    gp[64 + c] = (_Float16)p2;
    gp[96 + c] = (_Float16)p3;
  }
  __syncthreads();

  // ---- implicit GEMM: M=32 pixels (2 rows per wave), N=32, K=1152 ----
  const int lane = tid & 31;
  const int wave = tid >> 5;      // 0..7 -> rows 2w, 2w+1 within tile
  const int hs   = lane >> 4;     // K-half select per WMMA A/B layout
  const int ml   = lane & 15;
  const int r0   = 2 * wave;

  v8f acc00 = {};                 // row r0,   out channels  0..15
  v8f acc01 = {};                 // row r0,   out channels 16..31
  v8f acc10 = {};                 // row r0+1, out channels  0..15
  v8f acc11 = {};                 // row r0+1, out channels 16..31

  for (int tap = 0; tap < TAPS; ++tap) {
    const int dy = tap / 3;
    const int dx = tap - dy * 3;
    const _Float16* ga0 = sG + ((r0 + dy) * SROW + (ml + dx)) * GPAD + hs * 8;
    const _Float16* ga1 = ga0 + SROW * GPAD;       // next output row
    const _Float16* wb0 = sW + ml * WPAD + tap * GCH + hs * 8;
    const _Float16* wb1 = wb0 + 16 * WPAD;
#pragma unroll
    for (int kc = 0; kc < 4; ++kc) {
      V16u a0, a1, b0, b1;
      a0.h[0] = *(const v8h*)(ga0 + kc * 32);
      a0.h[1] = *(const v8h*)(ga0 + kc * 32 + 16);
      a1.h[0] = *(const v8h*)(ga1 + kc * 32);
      a1.h[1] = *(const v8h*)(ga1 + kc * 32 + 16);
      b0.h[0] = *(const v8h*)(wb0 + kc * 32);
      b0.h[1] = *(const v8h*)(wb0 + kc * 32 + 16);
      b1.h[0] = *(const v8h*)(wb1 + kc * 32);
      b1.h[1] = *(const v8h*)(wb1 + kc * 32 + 16);
      acc00 = __builtin_amdgcn_wmma_f32_16x16x32_f16(
          false, a0.v, false, b0.v, (short)0, acc00, false, false);
      acc01 = __builtin_amdgcn_wmma_f32_16x16x32_f16(
          false, a0.v, false, b1.v, (short)0, acc01, false, false);
      acc10 = __builtin_amdgcn_wmma_f32_16x16x32_f16(
          false, a1.v, false, b0.v, (short)0, acc10, false, false);
      acc11 = __builtin_amdgcn_wmma_f32_16x16x32_f16(
          false, a1.v, false, b1.v, (short)0, acc11, false, false);
    }
  }

  // ---- epilogue: scale by gate sum, write 2x float4 per C tile per lane ----
  const float s  = scale[b];
  const int row0 = ty0 + r0;
  const int col0 = tx0 + hs * 8;    // lane holds 8 consecutive output columns
  {
    float* po = out + (((size_t)b * 32 + ml) * HW + row0) * HW + col0;
    *(float4*)(po)     = make_float4(acc00[0] * s, acc00[1] * s, acc00[2] * s, acc00[3] * s);
    *(float4*)(po + 4) = make_float4(acc00[4] * s, acc00[5] * s, acc00[6] * s, acc00[7] * s);
    po += HW;                        // row0 + 1
    *(float4*)(po)     = make_float4(acc10[0] * s, acc10[1] * s, acc10[2] * s, acc10[3] * s);
    *(float4*)(po + 4) = make_float4(acc10[4] * s, acc10[5] * s, acc10[6] * s, acc10[7] * s);
  }
  {
    float* po = out + (((size_t)b * 32 + 16 + ml) * HW + row0) * HW + col0;
    *(float4*)(po)     = make_float4(acc01[0] * s, acc01[1] * s, acc01[2] * s, acc01[3] * s);
    *(float4*)(po + 4) = make_float4(acc01[4] * s, acc01[5] * s, acc01[6] * s, acc01[7] * s);
    po += HW;
    *(float4*)(po)     = make_float4(acc11[0] * s, acc11[1] * s, acc11[2] * s, acc11[3] * s);
    *(float4*)(po + 4) = make_float4(acc11[4] * s, acc11[5] * s, acc11[6] * s, acc11[7] * s);
  }
}

// ---------------------------------------------------------------------------
extern "C" void kernel_launch(void* const* d_in, const int* in_sizes, int n_in,
                              void* d_out, int out_size, void* d_ws, size_t ws_size,
                              hipStream_t stream) {
  const float* x     = (const float*)d_in[0];   // (16,32,192,192)
  const float* wgate = (const float*)d_in[1];   // (32,8)
  const float* poly  = (const float*)d_in[2];   // (1,32,128,3,3)
  const float* betaw = (const float*)d_in[3];   // (4,)
  float* out = (float*)d_out;                   // y flat + loss
  float* wsf = (float*)d_ws;
  float* gmean = wsf;                           // 512 floats
  float* scale = wsf + 512;                     // 16 floats
  float* loss  = out + (size_t)BATCH * CIN * HW * HW;

  kagn_colmean_kernel<<<dim3(CIN, BATCH), 256, 0, stream>>>(x, gmean);
  kagn_gate_kernel<<<1, 32, 0, stream>>>(gmean, wgate, scale, loss);
  kagn_conv_kernel<<<dim3(HW / TILE, HW / TILE, BATCH), NTHREADS, 0, stream>>>(
      x, poly, betaw, scale, out);
}
